// GGNNObj_35862976921807
// MI455X (gfx1250) — matrix operand order repaired and checked by reading, not compiled, not run on hardware
//
#include <hip/hip_runtime.h>
#include <math.h>

// ---------------- problem constants ----------------
#define NN    384
#define CC    151
#define HH    512
#define OUTD  512
#define TT    3
#define NC    (NN * CC)          // 57984 = 453 * 128
#define K2H   (2 * HH)           // 1024
#define CLK   (CC * OUTD)        // 77312 = 2416 * 32
#define CPAD  160                // C padded to multiple of 32 (K) / 16 (M)
#define CLNP  256                // classifier N padded to 2 * 128

#define USE_ASYNC_LDS 1

// ---------------- WMMA types ----------------
typedef __attribute__((ext_vector_type(16))) __bf16 v16bf;
typedef __attribute__((ext_vector_type(8)))  float  v8f;

union FragAB {
    v16bf v;
    unsigned int   u[8];
    unsigned short s[16];
};

__device__ __forceinline__ unsigned short f2bf(float f) {
    unsigned int u = __float_as_uint(f);
    u = (u + 0x7FFFu + ((u >> 16) & 1u)) >> 16;  // round-to-nearest-even
    return (unsigned short)u;
}

__device__ __forceinline__ v8f vzero8() {
    v8f z;
#pragma unroll
    for (int e = 0; e < 8; ++e) z[e] = 0.0f;
    return z;
}

// ---------------- CDNA5 async global->LDS helpers ----------------
__device__ __forceinline__ void async_ld_b128(const void* gaddr, unsigned ldsAddr) {
#if USE_ASYNC_LDS
    // GV mode: LDS address taken from the first vgpr, 64-bit global address from the second
    asm volatile("global_load_async_to_lds_b128 %0, %1, off"
                 :: "v"(ldsAddr), "v"(gaddr)
                 : "memory");
#endif
}

__device__ __forceinline__ void wait_async0() {
#if __has_builtin(__builtin_amdgcn_s_wait_asynccnt)
    __builtin_amdgcn_s_wait_asynccnt(0);
#else
    asm volatile("s_wait_asynccnt 0" ::: "memory");
#endif
}

// =====================================================================
// Generic bf16 GEMM:  C[M,N] (f32) = A[M,K] (bf16, row-major, lda)
//                                  x B[N,K]^T (bf16, row-major, ldb)
// Block: 256 threads = 8 waves; tile 128x128, K-step 32.
// Wave (wm in 0..3, wn in 0..1) owns a 32x64 subtile = 2x4 WMMA accums.
// Double-buffered LDS, filled by global_load_async_to_lds_b128 (ASYNCcnt).
// mode: 0 = store, 1 = accumulate (+=), 2 = atomicAdd (split-K)
// grid = (M/128, N/128, splits); each split handles kChunkSteps K-steps.
// =====================================================================
#define LDS_STR 40   // padded LDS row stride (bf16 elems) -> conflict-free

__global__ __launch_bounds__(256)
void k_gemm(const unsigned short* __restrict__ A, int lda,
            const unsigned short* __restrict__ B, int ldb,
            float* __restrict__ C, int ldc,
            int K, int kChunkSteps, int mode)
{
    __shared__ unsigned short As[2][128 * LDS_STR];
    __shared__ unsigned short Bs[2][128 * LDS_STR];

    const int tid  = threadIdx.x;
    const int lane = tid & 31;
    const int wave = tid >> 5;
    const int wm   = wave & 3;   // 4 waves along M  -> 4*32 = 128
    const int wn   = wave >> 2;  // 2 waves along N  -> 2*64 = 128
    const size_t mBase = (size_t)blockIdx.x * 128;
    const size_t nBase = (size_t)blockIdx.y * 128;

    v8f acc[2][4];
#pragma unroll
    for (int i = 0; i < 2; ++i)
#pragma unroll
        for (int j = 0; j < 4; ++j) acc[i][j] = vzero8();

    const int stepsTotal = K >> 5;
    int s0   = blockIdx.z * kChunkSteps;
    int sEnd = s0 + kChunkSteps;
    if (sEnd > stepsTotal) sEnd = stepsTotal;

    const int mrow = lane & 15;
    const int half = lane >> 4;

    // this thread's two 16B staging chunks: chunk = tid*2 + i
    const int r0 = (tid * 2) >> 2;          // row of chunk 0
    const int c0 = (tid * 2) & 3;           // 16B-chunk-in-row of chunk 0
    const int r1 = (tid * 2 + 1) >> 2;
    const int c1 = (tid * 2 + 1) & 3;

    // stage K-step s into LDS buffer b
    auto stage = [&](int s, int b) {
        const int k0 = s << 5;
#if USE_ASYNC_LDS
        async_ld_b128(A + (mBase + r0) * (size_t)lda + k0 + c0 * 8,
                      (unsigned)(size_t)&As[b][r0 * LDS_STR + c0 * 8]);
        async_ld_b128(B + (nBase + r0) * (size_t)ldb + k0 + c0 * 8,
                      (unsigned)(size_t)&Bs[b][r0 * LDS_STR + c0 * 8]);
        async_ld_b128(A + (mBase + r1) * (size_t)lda + k0 + c1 * 8,
                      (unsigned)(size_t)&As[b][r1 * LDS_STR + c1 * 8]);
        async_ld_b128(B + (nBase + r1) * (size_t)ldb + k0 + c1 * 8,
                      (unsigned)(size_t)&Bs[b][r1 * LDS_STR + c1 * 8]);
#else
        // fallback: issue all loads into distinct temps, then store (no per-chunk wait)
        const uint4 ta0 = *reinterpret_cast<const uint4*>(A + (mBase + r0) * (size_t)lda + k0 + c0 * 8);
        const uint4 tb0 = *reinterpret_cast<const uint4*>(B + (nBase + r0) * (size_t)ldb + k0 + c0 * 8);
        const uint4 ta1 = *reinterpret_cast<const uint4*>(A + (mBase + r1) * (size_t)lda + k0 + c1 * 8);
        const uint4 tb1 = *reinterpret_cast<const uint4*>(B + (nBase + r1) * (size_t)ldb + k0 + c1 * 8);
        *reinterpret_cast<uint4*>(&As[b][r0 * LDS_STR + c0 * 8]) = ta0;
        *reinterpret_cast<uint4*>(&Bs[b][r0 * LDS_STR + c0 * 8]) = tb0;
        *reinterpret_cast<uint4*>(&As[b][r1 * LDS_STR + c1 * 8]) = ta1;
        *reinterpret_cast<uint4*>(&Bs[b][r1 * LDS_STR + c1 * 8]) = tb1;
#endif
    };

    // prologue: fill buffer 0
    stage(s0, 0);
    wait_async0();
    __syncthreads();

    for (int s = s0; s < sEnd; ++s) {
        const int cur = (s - s0) & 1;
        // kick off async fill of the other buffer while we compute
        if (s + 1 < sEnd) stage(s + 1, cur ^ 1);

        // ---- build fragments from LDS buffer `cur`
        FragAB fa[2], fb[4];
#pragma unroll
        for (int mi = 0; mi < 2; ++mi) {
            const int rb = wm * 32 + mi * 16 + mrow;
#pragma unroll
            for (int v = 0; v < 8; ++v) {
                const int kk = ((v & 4) ? 16 : 0) + half * 8 + (v & 3) * 2;
                fa[mi].u[v] = *reinterpret_cast<const unsigned int*>(&As[cur][rb * LDS_STR + kk]);
            }
        }
#pragma unroll
        for (int ni = 0; ni < 4; ++ni) {
            const int cb = wn * 64 + ni * 16 + mrow;
#pragma unroll
            for (int v = 0; v < 8; ++v) {
                const int kk = ((v & 4) ? 16 : 0) + half * 8 + (v & 3) * 2;
                fb[ni].u[v] = *reinterpret_cast<const unsigned int*>(&Bs[cur][cb * LDS_STR + kk]);
            }
        }
        // ---- 8 WMMAs
#pragma unroll
        for (int mi = 0; mi < 2; ++mi)
#pragma unroll
            for (int ni = 0; ni < 4; ++ni)
                acc[mi][ni] = __builtin_amdgcn_wmma_f32_16x16x32_bf16(
                    false, fa[mi].v, false, fb[ni].v,
                    (short)0, acc[mi][ni], false, false);

        // next buffer must be fully written before anyone reads it, and this
        // buffer must be fully read (our ds reads completed before our wmma)
        // before iteration s+2 overwrites it.
        wait_async0();
        __syncthreads();
    }

    // ---- epilogue
    const int col0 = lane & 15;
    const int rofs = half * 8;
#pragma unroll
    for (int mi = 0; mi < 2; ++mi)
#pragma unroll
        for (int ni = 0; ni < 4; ++ni) {
            const size_t row0 = mBase + wm * 32 + mi * 16 + rofs;
            const size_t col  = nBase + wn * 64 + ni * 16 + col0;
#pragma unroll
            for (int g = 0; g < 8; ++g) {
                float val = acc[mi][ni][g];
                float* p  = C + (row0 + g) * (size_t)ldc + col;
                if (mode == 0)      *p = val;
                else if (mode == 1) *p += val;
                else                atomicAdd(p, val);
            }
        }
}

// =====================================================================
// av kernel: per sample n, av1 = M1 - matT @ hidden[n], av2 = M2 - mat @ hidden[n]
// One wave computes a 16(a) x 16(h) output tile, K = 160 (padded C), 5 WMMAs x2.
// =====================================================================
__global__ __launch_bounds__(256)
void k_av(const unsigned short* __restrict__ matT,   // [160,160] bf16 (a,k)
          const unsigned short* __restrict__ matR,   // [160,160] bf16 (a,k)
          const unsigned short* __restrict__ hbf,    // [NC, H] bf16
          const float* __restrict__ M1,              // [C,H]
          const float* __restrict__ M2,              // [C,H]
          unsigned short* __restrict__ av)           // [NC, 2H] bf16
{
    const int lane = threadIdx.x & 31;
    const int wg   = blockIdx.x * 8 + (threadIdx.x >> 5);
    const int aT   = wg % 10;             // 10 a-tiles (160/16)
    const int hT   = (wg / 10) % 32;      // 32 h-tiles (512/16)
    const int n    = wg / 320;            // sample
    const int mrow = lane & 15;
    const int half = lane >> 4;
    const int col  = hT * 16 + mrow;

    v8f acc1 = vzero8(), acc2 = vzero8();

#pragma unroll
    for (int kt = 0; kt < 5; ++kt) {
        const int k0 = kt * 32;
        FragAB a1, a2, b;
        const int rb = aT * 16 + mrow;
#pragma unroll
        for (int v = 0; v < 8; ++v) {
            const int kk = k0 + ((v & 4) ? 16 : 0) + half * 8 + (v & 3) * 2;
            a1.u[v] = *reinterpret_cast<const unsigned int*>(&matT[rb * CPAD + kk]);
            a2.u[v] = *reinterpret_cast<const unsigned int*>(&matR[rb * CPAD + kk]);
        }
#pragma unroll
        for (int v = 0; v < 8; ++v) {
            const int k = k0 + ((v & 4) ? 16 : 0) + half * 8 + (v & 3) * 2;
            unsigned short lo = (k     < CC) ? hbf[((size_t)n * CC + k    ) * HH + col] : (unsigned short)0;
            unsigned short hi = (k + 1 < CC) ? hbf[((size_t)n * CC + k + 1) * HH + col] : (unsigned short)0;
            b.s[2 * v]     = lo;
            b.s[2 * v + 1] = hi;
        }
        acc1 = __builtin_amdgcn_wmma_f32_16x16x32_bf16(false, a1.v, false, b.v, (short)0, acc1, false, false);
        acc2 = __builtin_amdgcn_wmma_f32_16x16x32_bf16(false, a2.v, false, b.v, (short)0, acc2, false, false);
    }

#pragma unroll
    for (int g = 0; g < 8; ++g) {
        const int a = aT * 16 + g + half * 8;
        if (a < CC) {
            const size_t r = (size_t)n * CC + a;
            const float m1v = M1[a * HH + col];
            const float m2v = M2[a * HH + col];
            av[r * K2H + col]      = f2bf(m1v - acc1[g]);
            av[r * K2H + HH + col] = f2bf(m2v - acc2[g]);
        }
    }
}

// ---------------- small / elementwise kernels ----------------
__global__ void k_init(const float* __restrict__ inp, float* __restrict__ hf32,
                       unsigned short* __restrict__ hbf) {
    const size_t idx = (size_t)blockIdx.x * 256 + threadIdx.x;
    if (idx >= (size_t)NC * HH) return;
    const size_t r = idx / HH;
    const int h = (int)(idx % HH);
    const int n = (int)(r / CC);
    const float v = inp[(size_t)n * HH + h];
    hf32[idx] = v;
    hbf[idx]  = f2bf(v);
}

__global__ void k_hidden_sum(const float* __restrict__ hf32, float* __restrict__ hsum) {
    const int idx = blockIdx.x * 256 + threadIdx.x;
    if (idx >= CC * HH) return;
    const int c = idx / HH, h = idx % HH;
    float s = 0.f;
    for (int n = 0; n < NN; ++n) s += hf32[((size_t)n * CC + c) * HH + h];
    hsum[idx] = s;
}

__global__ void k_m1m2(const float* __restrict__ matrix, const float* __restrict__ hsum,
                       float* __restrict__ M1, float* __restrict__ M2) {
    const int idx = blockIdx.x * 256 + threadIdx.x;
    if (idx >= CC * HH) return;
    const int a = idx / HH, h = idx % HH;
    float s1 = 0.f, s2 = 0.f;
    for (int b = 0; b < CC; ++b) {
        const float hv = hsum[b * HH + h];
        s1 = fmaf(matrix[b * CC + a], hv, s1);
        s2 = fmaf(matrix[a * CC + b], hv, s2);
    }
    M1[idx] = s1;
    M2[idx] = s2;
}

// z = sigmoid(zpre), r = sigmoid(rpre), rh = bf16(r * h)
__global__ void k_elem1(float* __restrict__ czr, const float* __restrict__ hf32,
                        const float* __restrict__ w3b, const float* __restrict__ u3b,
                        const float* __restrict__ w4b, const float* __restrict__ u4b,
                        unsigned short* __restrict__ rhb) {
    const size_t idx = (size_t)blockIdx.x * 256 + threadIdx.x;
    if (idx >= (size_t)NC * HH) return;
    const size_t r = idx / HH;
    const int h = (int)(idx % HH);
    const float zpre = czr[r * K2H + h]      + w3b[h] + u3b[h];
    const float rpre = czr[r * K2H + HH + h] + w4b[h] + u4b[h];
    const float z  = 1.f / (1.f + expf(-zpre));
    const float rv = 1.f / (1.f + expf(-rpre));
    czr[r * K2H + h] = z;                  // keep z for the update step
    rhb[idx] = f2bf(rv * hf32[idx]);
}

// hv = tanh(pre), h = (1-z)h + z*hv ; refresh bf16 copy
__global__ void k_elem2(const float* __restrict__ chv, float* __restrict__ czr,
                        const float* __restrict__ w5b, const float* __restrict__ u5b,
                        float* __restrict__ hf32, unsigned short* __restrict__ hbf) {
    const size_t idx = (size_t)blockIdx.x * 256 + threadIdx.x;
    if (idx >= (size_t)NC * HH) return;
    const size_t r = idx / HH;
    const int h = (int)(idx % HH);
    const float hv = tanhf(chv[idx] + w5b[h] + u5b[h]);
    const float z  = czr[r * K2H + h];
    const float nh = (1.f - z) * hf32[idx] + z * hv;
    hf32[idx] = nh;
    hbf[idx]  = f2bf(nh);
}

// outin = [ bf16(hidden) | bf16(input broadcast) ]  (reuses av buffer)
__global__ void k_outin(const unsigned short* __restrict__ hbf,
                        const float* __restrict__ inp,
                        unsigned short* __restrict__ outin) {
    const size_t idx = (size_t)blockIdx.x * 256 + threadIdx.x;
    if (idx >= (size_t)NC * K2H) return;
    const size_t r = idx >> 10;
    const int c = (int)(idx & 1023);
    outin[idx] = (c < HH) ? hbf[r * HH + c]
                          : f2bf(inp[(r / CC) * (size_t)HH + (c - HH)]);
}

__global__ void k_relu_bf(const float* __restrict__ chv, const float* __restrict__ outb,
                          unsigned short* __restrict__ dst) {
    const size_t idx = (size_t)blockIdx.x * 256 + threadIdx.x;
    if (idx >= (size_t)NC * OUTD) return;
    const int o = (int)(idx % OUTD);
    dst[idx] = f2bf(fmaxf(chv[idx] + outb[o], 0.f));
}

__global__ void k_zero(float* __restrict__ p, int n) {
    const int idx = blockIdx.x * 256 + threadIdx.x;
    if (idx < n) p[idx] = 0.f;
}

__global__ void k_bias_out(const float* __restrict__ tmp, const float* __restrict__ clsb,
                           float* __restrict__ out) {
    const int idx = blockIdx.x * 256 + threadIdx.x;
    if (idx >= NN * CC) return;
    const int n = idx / CC, a = idx % CC;
    out[idx] = tmp[n * CLNP + a] + clsb[a];
}

// ---------------- weight conversion kernels ----------------
__global__ void k_cvt_pair(unsigned short* __restrict__ dst, const float* __restrict__ s0,
                           const float* __restrict__ s1, int halfRows, int Kd) {
    const size_t idx = (size_t)blockIdx.x * 256 + threadIdx.x;
    const size_t total = (size_t)2 * halfRows * Kd;
    if (idx >= total) return;
    const size_t j = idx / Kd;
    dst[idx] = f2bf(j < (size_t)halfRows ? s0[idx] : s1[idx - (size_t)halfRows * Kd]);
}

__global__ void k_cvt_simple(unsigned short* __restrict__ dst, const float* __restrict__ src,
                             size_t n) {
    const size_t idx = (size_t)blockIdx.x * 256 + threadIdx.x;
    if (idx < n) dst[idx] = f2bf(src[idx]);
}

__global__ void k_cvt_cls(unsigned short* __restrict__ dst, const float* __restrict__ src) {
    const size_t idx = (size_t)blockIdx.x * 256 + threadIdx.x;
    if (idx >= (size_t)CLNP * CLK) return;
    dst[idx] = (idx < (size_t)CC * CLK) ? f2bf(src[idx]) : (unsigned short)0;
}

__global__ void k_cvt_matrix(const float* __restrict__ matrix,
                             unsigned short* __restrict__ matT,
                             unsigned short* __restrict__ matR) {
    const int idx = blockIdx.x * 256 + threadIdx.x;
    if (idx >= CPAD * CPAD) return;
    const int a = idx / CPAD, b = idx % CPAD;
    const bool in = (a < CC) && (b < CC);
    matT[idx] = in ? f2bf(matrix[b * CC + a]) : (unsigned short)0;
    matR[idx] = in ? f2bf(matrix[a * CC + b]) : (unsigned short)0;
}

// ---------------- host launcher ----------------
static inline size_t alignUp(size_t x) { return (x + 255) & ~(size_t)255; }
static inline unsigned int cdiv(size_t a, size_t b) { return (unsigned int)((a + b - 1) / b); }

extern "C" void kernel_launch(void* const* d_in, const int* in_sizes, int n_in,
                              void* d_out, int out_size, void* d_ws, size_t ws_size,
                              hipStream_t stream) {
    (void)in_sizes; (void)n_in; (void)out_size; (void)ws_size;

    const float* inp    = (const float*)d_in[0];
    const float* matrix = (const float*)d_in[1];
    const float* w3w = (const float*)d_in[2];  const float* w3b = (const float*)d_in[3];
    const float* u3w = (const float*)d_in[4];  const float* u3b = (const float*)d_in[5];
    const float* w4w = (const float*)d_in[6];  const float* w4b = (const float*)d_in[7];
    const float* u4w = (const float*)d_in[8];  const float* u4b = (const float*)d_in[9];
    const float* w5w = (const float*)d_in[10]; const float* w5b = (const float*)d_in[11];
    const float* u5w = (const float*)d_in[12]; const float* u5b = (const float*)d_in[13];
    const float* outw = (const float*)d_in[14]; const float* outb = (const float*)d_in[15];
    const float* clsw = (const float*)d_in[16]; const float* clsb = (const float*)d_in[17];
    float* out = (float*)d_out;

    // workspace carve-up
    char* ws = (char*)d_ws;
    size_t off = 0;
    auto take = [&](size_t bytes) { char* p = ws + off; off += alignUp(bytes); return p; };

    float*          hf32  = (float*)         take((size_t)NC * HH * 4);
    unsigned short* hbf   = (unsigned short*)take((size_t)NC * HH * 2);
    unsigned short* avb   = (unsigned short*)take((size_t)NC * K2H * 2);   // av / outin
    float*          czr   = (float*)         take((size_t)NC * K2H * 4);   // z,r pre / relu-bf16
    float*          chv   = (float*)         take((size_t)NC * HH * 4);    // hv pre / out pre
    unsigned short* rhb   = (unsigned short*)take((size_t)NC * HH * 2);
    float*          hsum  = (float*)         take((size_t)CC * HH * 4);
    float*          M1    = (float*)         take((size_t)CC * HH * 4);
    float*          M2    = (float*)         take((size_t)CC * HH * 4);
    unsigned short* matT  = (unsigned short*)take((size_t)CPAD * CPAD * 2);
    unsigned short* matR  = (unsigned short*)take((size_t)CPAD * CPAD * 2);
    unsigned short* wzr   = (unsigned short*)take((size_t)K2H * K2H * 2);  // [w3;w4], N=1024,K=1024
    unsigned short* w5bf  = (unsigned short*)take((size_t)HH * K2H * 2);
    unsigned short* uzr   = (unsigned short*)take((size_t)K2H * HH * 2);   // [u3;u4], N=1024,K=512
    unsigned short* u5bf  = (unsigned short*)take((size_t)HH * HH * 2);
    unsigned short* outwb = (unsigned short*)take((size_t)OUTD * K2H * 2);
    unsigned short* clswb = (unsigned short*)take((size_t)CLNP * CLK * 2);
    float*          ctmp  = (float*)         take((size_t)NN * CLNP * 4);

    const size_t ncH  = (size_t)NC * HH;    // 29,687,808
    const size_t nc2H = (size_t)NC * K2H;   // 59,375,616

    // ---- weight conversion (cheap, deterministic every call)
    k_cvt_pair  <<<cdiv((size_t)K2H * K2H, 256), 256, 0, stream>>>(wzr, w3w, w4w, HH, K2H);
    k_cvt_pair  <<<cdiv((size_t)K2H * HH, 256), 256, 0, stream>>>(uzr, u3w, u4w, HH, HH);
    k_cvt_simple<<<cdiv((size_t)HH * K2H, 256), 256, 0, stream>>>(w5bf, w5w, (size_t)HH * K2H);
    k_cvt_simple<<<cdiv((size_t)HH * HH, 256), 256, 0, stream>>>(u5bf, u5w, (size_t)HH * HH);
    k_cvt_simple<<<cdiv((size_t)OUTD * K2H, 256), 256, 0, stream>>>(outwb, outw, (size_t)OUTD * K2H);
    k_cvt_cls   <<<cdiv((size_t)CLNP * CLK, 256), 256, 0, stream>>>(clswb, clsw);
    k_cvt_matrix<<<cdiv((size_t)CPAD * CPAD, 256), 256, 0, stream>>>(matrix, matT, matR);

    // ---- init hidden
    k_init<<<cdiv(ncH, 256), 256, 0, stream>>>(inp, hf32, hbf);

    const dim3 blk(256);
    for (int t = 0; t < TT; ++t) {
        k_hidden_sum<<<cdiv((size_t)CC * HH, 256), blk, 0, stream>>>(hf32, hsum);
        k_m1m2      <<<cdiv((size_t)CC * HH, 256), blk, 0, stream>>>(matrix, hsum, M1, M2);
        // av (batched WMMA): 384 * 10 * 32 waves / 8 per block
        k_av<<<dim3(NN * 10 * 32 / 8), blk, 0, stream>>>(matT, matR, hbf, M1, M2, avb);
        // z,r pre-activations: czr = av @ [w3;w4]^T + hbf @ [u3;u4]^T
        k_gemm<<<dim3(NC / 128, K2H / 128, 1), blk, 0, stream>>>(avb, K2H, wzr, K2H, czr, K2H,
                                                                 K2H, K2H / 32, 0);
        k_gemm<<<dim3(NC / 128, K2H / 128, 1), blk, 0, stream>>>(hbf, HH, uzr, HH, czr, K2H,
                                                                 HH, HH / 32, 1);
        // hv pre (part 1): chv = av @ w5^T
        k_gemm<<<dim3(NC / 128, HH / 128, 1), blk, 0, stream>>>(avb, K2H, w5bf, K2H, chv, HH,
                                                                K2H, K2H / 32, 0);
        k_elem1<<<cdiv(ncH, 256), blk, 0, stream>>>(czr, hf32, w3b, u3b, w4b, u4b, rhb);
        // hv pre (part 2): chv += (r*h) @ u5^T
        k_gemm<<<dim3(NC / 128, HH / 128, 1), blk, 0, stream>>>(rhb, HH, u5bf, HH, chv, HH,
                                                                HH, HH / 32, 1);
        k_elem2<<<cdiv(ncH, 256), blk, 0, stream>>>(chv, czr, w5b, u5b, hf32, hbf);
    }

    // ---- output projection: outin = [h | input], chv = outin @ out_w^T
    k_outin<<<cdiv(nc2H, 256), blk, 0, stream>>>(hbf, inp, avb);
    k_gemm<<<dim3(NC / 128, OUTD / 128, 1), blk, 0, stream>>>(avb, K2H, outwb, K2H, chv, OUTD,
                                                              K2H, K2H / 32, 0);
    // relu -> bf16 (into czr buffer, viewed as [N, C*OUT] bf16)
    unsigned short* reluBf = (unsigned short*)czr;
    k_relu_bf<<<cdiv((size_t)NC * OUTD, 256), blk, 0, stream>>>(chv, outb, reluBf);

    // ---- classifier: split-K=16 atomic GEMM over K=77312
    k_zero<<<cdiv((size_t)NN * CLNP, 256), blk, 0, stream>>>(ctmp, NN * CLNP);
    k_gemm<<<dim3(NN / 128, CLNP / 128, 16), blk, 0, stream>>>(reluBf, CLK, clswb, CLK,
                                                               ctmp, CLNP, CLK,
                                                               (CLK / 32) / 16, 2);
    k_bias_out<<<cdiv((size_t)NN * CC, 256), blk, 0, stream>>>(ctmp, clsb, out);
}
